// HiPPODiscretization_3_81913616270051
// MI455X (gfx1250) — matrix operand order repaired and checked by reading, not compiled, not run on hardware
//
#include <hip/hip_runtime.h>

// ---------------------------------------------------------------- constants
#define LSEQ    2048
#define DMODEL  1024
#define DINNER  1024
#define DSTATE  16
#define DTRANK  64
#define DCONV   4
#define XPROJ   (DTRANK + 2 * DSTATE)   // 96

#define CHUNK   64
#define NCHUNK  (LSEQ / CHUNK)          // 32

// LDS GEMM tiling
#define BM 64
#define BN 128
#define BK 32

// ---------------------------------------------------------------- types
typedef __attribute__((ext_vector_type(16))) __bf16 bf16x16;
typedef __attribute__((ext_vector_type(8)))  float  v8f;
typedef __attribute__((ext_vector_type(4)))  float  f32x4;

union FragU { bf16x16 v; struct { f32x4 a, b; } q; };

// fp32 -> bf16 (round to nearest even), stored as raw u16
__device__ __forceinline__ unsigned short f2bf(float f) {
    unsigned int u = __float_as_uint(f);
    u = (u + 0x7FFFu + ((u >> 16) & 1u)) >> 16;
    return (unsigned short)u;
}

// ---------------------------------------------------------------- fragment loads
// A fragment: 16x32 bf16.  ISA 7.12.2: lanes 0-15 = M, h=lane>>4 selects
// K runs [8h,8h+8) and [16+8h,16+8h+8)
__device__ __forceinline__ bf16x16 frag_a(const unsigned short* base, int ld,
                                          int row0, int k0, int lane) {
    int r = lane & 15, h = lane >> 4;
    const unsigned short* p = base + (long)(row0 + r) * ld + k0 + 8 * h;
    FragU u;
    u.q.a = *(const f32x4*)(p);
    u.q.b = *(const f32x4*)(p + 16);
    return u.v;
}
// B fragment: 32x16 bf16, source stored [N][K]; lanes 0-15 = N,
// h selects K runs [8h,8h+8) and [16+8h,16+8h+8)
__device__ __forceinline__ bf16x16 frag_b(const unsigned short* base, int ld,
                                          int col0, int k0, int lane) {
    int n = lane & 15, h = lane >> 4;
    const unsigned short* p = base + (long)(col0 + n) * ld + k0 + 8 * h;
    FragU u;
    u.q.a = *(const f32x4*)(p);
    u.q.b = *(const f32x4*)(p + 16);
    return u.v;
}

// ---------------------------------------------------------------- async copy
// 16B global -> LDS via the CDNA5 async path (tracked by ASYNCcnt).
// The LDS address operand is derived from the real shared-memory pointer:
// ISA: flat addr of LDS = {aperture, offset[31:0]}, so the low 32 bits are
// exactly the LDS byte offset the instruction's VDST expects.  The ptrtoint
// also makes the shared array escape, so the compiler keeps the ds_loads and
// treats the asm as a potential writer of LDS.
__device__ __forceinline__ void async_copy16(unsigned short* lds,
                                             const unsigned short* g) {
    unsigned loff = (unsigned)(unsigned long long)lds;
    asm volatile("global_load_async_to_lds_b128 %0, %1, off"
                 :: "v"(loff), "v"((unsigned long long)g) : "memory");
}

// ---------------------------------------------------------------- GEMM A: LDS-staged
// C[M,N] = A[M,K] * B[N,K]^T.  Block = 256 threads = 8 waves (2 M x 4 N),
// block tile 64x128, K-step 32, double-buffered LDS filled with
// global_load_async_to_lds_b128.  Requires M%64==0, N%128==0, K%32==0.
#define SLAB_ELEMS (BM * BK + BN * BK)            // 6144 u16 per buffer

__device__ __forceinline__ void async_slab(unsigned short* lds,   // buffer base
                                           const unsigned short* A,
                                           const unsigned short* B,
                                           int K, long rowA0, long rowB0,
                                           int k, int t) {
    // A slab: 64 rows x 32 cols = 256 chunks of 16B, one per thread
    {
        int row = t >> 2, seg = t & 3;
        async_copy16(lds + row * BK + seg * 8,
                     A + (rowA0 + row) * (long)K + k + seg * 8);
    }
    // B slab: 128 rows x 32 cols = 512 chunks, two per thread
#pragma unroll
    for (int i = 0; i < 2; ++i) {
        int idx = t + i * 256;
        int row = idx >> 2, seg = idx & 3;
        async_copy16(lds + BM * BK + row * BK + seg * 8,
                     B + (rowB0 + row) * (long)K + k + seg * 8);
    }
}

__device__ __forceinline__ bf16x16 lds_frag_a(const unsigned short* s, int row0, int lane) {
    int r = lane & 15, h = lane >> 4;
    const unsigned short* p = s + (row0 + r) * BK + 8 * h;
    FragU u;
    u.q.a = *(const f32x4*)(p);
    u.q.b = *(const f32x4*)(p + 16);
    return u.v;
}
__device__ __forceinline__ bf16x16 lds_frag_b(const unsigned short* s, int col0, int lane) {
    int n = lane & 15, h = lane >> 4;
    const unsigned short* p = s + (col0 + n) * BK + 8 * h;
    FragU u;
    u.q.a = *(const f32x4*)(p);
    u.q.b = *(const f32x4*)(p + 16);
    return u.v;
}

__global__ void __launch_bounds__(256)
gemm_lds_wmma(const unsigned short* __restrict__ A,
              const unsigned short* __restrict__ B,
              float* __restrict__ C, int K, int ldc) {
    __shared__ __align__(16) unsigned short smem[2 * SLAB_ELEMS];

    int t    = threadIdx.x;
    int lane = t & 31;
    int w    = t >> 5;            // 0..7
    int wm   = w >> 2;            // 0..1
    int wn   = w & 3;             // 0..3
    long blockM = (long)blockIdx.y * BM;
    long blockN = (long)blockIdx.x * BN;

    async_slab(smem, A, B, K, blockM, blockN, 0, t);
    asm volatile("s_wait_asynccnt 0x0" ::: "memory");
    __syncthreads();

    v8f acc00 = {}, acc01 = {}, acc10 = {}, acc11 = {};
    unsigned cur = 0;

    for (int k = 0; k < K; k += BK) {
        if (k + BK < K)
            async_slab(smem + (cur ^ 1u) * SLAB_ELEMS, A, B, K, blockM, blockN, k + BK, t);

        const unsigned short* As = smem + cur * SLAB_ELEMS;
        const unsigned short* Bs = As + BM * BK;

        bf16x16 a0 = lds_frag_a(As, wm * 32,      lane);
        bf16x16 a1 = lds_frag_a(As, wm * 32 + 16, lane);
        bf16x16 b0 = lds_frag_b(Bs, wn * 32,      lane);
        bf16x16 b1 = lds_frag_b(Bs, wn * 32 + 16, lane);

        acc00 = __builtin_amdgcn_wmma_f32_16x16x32_bf16(false, a0, false, b0, (short)0, acc00, false, false);
        acc01 = __builtin_amdgcn_wmma_f32_16x16x32_bf16(false, a0, false, b1, (short)0, acc01, false, false);
        acc10 = __builtin_amdgcn_wmma_f32_16x16x32_bf16(false, a1, false, b0, (short)0, acc10, false, false);
        acc11 = __builtin_amdgcn_wmma_f32_16x16x32_bf16(false, a1, false, b1, (short)0, acc11, false, false);

        asm volatile("s_wait_asynccnt 0x0" ::: "memory");
        __syncthreads();
        cur ^= 1u;
    }

    long m0 = blockM + wm * 32;
    long n0 = blockN + wn * 32;
    int  n  = lane & 15;
    int  mo = (lane >> 4) << 3;
#pragma unroll
    for (int r = 0; r < 8; ++r) {
        long mA = m0 + mo + r;
        long mB = m0 + 16 + mo + r;
        C[mA * ldc + n0 +      n] = acc00[r];
        C[mA * ldc + n0 + 16 + n] = acc01[r];
        C[mB * ldc + n0 +      n] = acc10[r];
        C[mB * ldc + n0 + 16 + n] = acc11[r];
    }
}

// ---------------------------------------------------------------- GEMM B: direct (odd shapes)
// One wave computes a 32x32 tile; M%32==0, N%32==0, K%32==0.
__global__ void gemm_wmma_bf16(const unsigned short* __restrict__ A,
                               const unsigned short* __restrict__ B,
                               float* __restrict__ C,
                               int M, int N, int K, int ldc) {
    int lane = threadIdx.x & 31;
    int wid  = (int)((blockIdx.x * blockDim.x + threadIdx.x) >> 5);
    int tilesN = N >> 5;
    int tilesM = M >> 5;
    if (wid >= tilesM * tilesN) return;          // wave-uniform, EXEC stays full
    int m0 = (wid / tilesN) << 5;
    int n0 = (wid % tilesN) << 5;

    v8f acc00 = {}, acc01 = {}, acc10 = {}, acc11 = {};

    for (int k = 0; k < K; k += 32) {
        if (k + 32 < K) {
            __builtin_prefetch(A + (long)(m0 + (lane & 15)) * K + k + 32, 0, 3);
            __builtin_prefetch(B + (long)(n0 + (lane & 15)) * K + k + 32, 0, 3);
        }
        bf16x16 a0 = frag_a(A, K, m0,      k, lane);
        bf16x16 a1 = frag_a(A, K, m0 + 16, k, lane);
        bf16x16 b0 = frag_b(B, K, n0,      k, lane);
        bf16x16 b1 = frag_b(B, K, n0 + 16, k, lane);
        acc00 = __builtin_amdgcn_wmma_f32_16x16x32_bf16(false, a0, false, b0, (short)0, acc00, false, false);
        acc01 = __builtin_amdgcn_wmma_f32_16x16x32_bf16(false, a0, false, b1, (short)0, acc01, false, false);
        acc10 = __builtin_amdgcn_wmma_f32_16x16x32_bf16(false, a1, false, b0, (short)0, acc10, false, false);
        acc11 = __builtin_amdgcn_wmma_f32_16x16x32_bf16(false, a1, false, b1, (short)0, acc11, false, false);
    }

    int n  = lane & 15;
    int mo = (lane >> 4) << 3;
#pragma unroll
    for (int r = 0; r < 8; ++r) {
        int mA = m0 + mo + r;
        int mB = m0 + 16 + mo + r;
        C[(long)mA * ldc + n0 +      n] = acc00[r];
        C[(long)mA * ldc + n0 + 16 + n] = acc01[r];
        C[(long)mB * ldc + n0 +      n] = acc10[r];
        C[(long)mB * ldc + n0 + 16 + n] = acc11[r];
    }
}

// ---------------------------------------------------------------- elementwise
__global__ void f32_to_bf16_kernel(const float* __restrict__ in,
                                   unsigned short* __restrict__ out, long n) {
    long i = (long)blockIdx.x * blockDim.x + threadIdx.x;
    if (i < n) out[i] = f2bf(in[i]);
}

// depthwise causal conv(4) + bias + SiLU over x = xz[:, :DINNER]
__global__ void conv_silu_kernel(const float* __restrict__ xz,
                                 const float* __restrict__ cw,
                                 const float* __restrict__ cb,
                                 float* __restrict__ xconv,
                                 unsigned short* __restrict__ xconv_bf) {
    long idx = (long)blockIdx.x * blockDim.x + threadIdx.x;
    if (idx >= (long)LSEQ * DINNER) return;
    int l = (int)(idx >> 10);
    int d = (int)(idx & (DINNER - 1));
    float acc = cb[d];
#pragma unroll
    for (int j = 0; j < DCONV; ++j) {
        int ls = l - (DCONV - 1) + j;
        if (ls >= 0) acc += cw[d * DCONV + j] * xz[(long)ls * (2 * DINNER) + d];
    }
    float s = acc / (1.f + __expf(-acc));        // SiLU
    xconv[idx]    = s;
    xconv_bf[idx] = f2bf(s);
}

// x_dbl[:, :DTRANK] -> bf16 (packed lda = DTRANK)
__global__ void extract_dt_kernel(const float* __restrict__ xdbl,
                                  unsigned short* __restrict__ dt_bf) {
    long idx = (long)blockIdx.x * blockDim.x + threadIdx.x;
    if (idx >= (long)LSEQ * DTRANK) return;
    int l = (int)(idx >> 6);
    int r = (int)(idx & (DTRANK - 1));
    dt_bf[idx] = f2bf(xdbl[(long)l * XPROJ + r]);
}

// delta = softplus(delta_pre + dt_bias)
__global__ void softplus_bias_kernel(float* __restrict__ delta,
                                     const float* __restrict__ dpb) {
    long idx = (long)blockIdx.x * blockDim.x + threadIdx.x;
    if (idx >= (long)LSEQ * DINNER) return;
    int d = (int)(idx & (DINNER - 1));
    float v = delta[idx] + dpb[d];
    delta[idx] = (v > 20.f) ? v : log1pf(__expf(v));
}

// ---------------------------------------------------------------- chunked scan
// h_l = dA_l * h_{l-1} + dBu_l is associative: (a,b)o(a',b') = (a*a', a'*b+b').
// Pass A: per (chunk, channel) compute P[n]=prod(dA), hend[n] with h_in=0.
__global__ void scan_chunk_local(const float* __restrict__ delta,
                                 const float* __restrict__ xconv,
                                 const float* __restrict__ xdbl,
                                 const float* __restrict__ A_log,
                                 float* __restrict__ Pout,      // [NCHUNK][DINNER][16]
                                 float* __restrict__ Hend) {    // [NCHUNK][DINNER][16]
    int gid = (int)(blockIdx.x * blockDim.x + threadIdx.x);
    if (gid >= NCHUNK * DINNER) return;
    int c = gid / DINNER;
    int d = gid % DINNER;

    float An[DSTATE], P[DSTATE], h[DSTATE];
#pragma unroll
    for (int n = 0; n < DSTATE; ++n) {
        An[n] = -__expf(A_log[d * DSTATE + n]);
        P[n] = 1.f; h[n] = 0.f;
    }

    for (int l = c * CHUNK; l < c * CHUNK + CHUNK; ++l) {
        float dlt = delta[(long)l * DINNER + d];
        float dx  = dlt * xconv[(long)l * DINNER + d];
        const f32x4* bc = (const f32x4*)(xdbl + (long)l * XPROJ + DTRANK);
        float Bv[DSTATE];
        ((f32x4*)Bv)[0] = bc[0]; ((f32x4*)Bv)[1] = bc[1];
        ((f32x4*)Bv)[2] = bc[2]; ((f32x4*)Bv)[3] = bc[3];
#pragma unroll
        for (int n = 0; n < DSTATE; ++n) {
            float dA = __expf(dlt * An[n]);
            P[n] *= dA;
            h[n]  = dA * h[n] + dx * Bv[n];
        }
    }
    long base = (long)gid * DSTATE;
#pragma unroll
    for (int n = 0; n < DSTATE; ++n) { Pout[base + n] = P[n]; Hend[base + n] = h[n]; }
}

// Pass B: serial prefix over the 32 chunk summaries, per (d, n) state.
__global__ void scan_chunk_prefix(const float* __restrict__ P,
                                  const float* __restrict__ Hend,
                                  float* __restrict__ Hin) {    // [NCHUNK][DINNER][16]
    int gid = (int)(blockIdx.x * blockDim.x + threadIdx.x);    // d*16+n
    if (gid >= DINNER * DSTATE) return;
    float h = 0.f;
    const long stride = (long)DINNER * DSTATE;
    for (int c = 0; c < NCHUNK; ++c) {
        Hin[c * stride + gid] = h;
        h = P[c * stride + gid] * h + Hend[c * stride + gid];
    }
}

// Pass C: replay each chunk seeded with Hin; emit gated y in bf16.
__global__ void scan_chunk_apply(const float* __restrict__ delta,
                                 const float* __restrict__ xconv,
                                 const float* __restrict__ xdbl,
                                 const float* __restrict__ xz,      // z at cols [DINNER,2*DINNER)
                                 const float* __restrict__ A_log,
                                 const float* __restrict__ Dp,
                                 const float* __restrict__ Hin,
                                 unsigned short* __restrict__ y_bf) {
    int gid = (int)(blockIdx.x * blockDim.x + threadIdx.x);
    if (gid >= NCHUNK * DINNER) return;
    int c = gid / DINNER;
    int d = gid % DINNER;

    float An[DSTATE], h[DSTATE];
    long base = (long)gid * DSTATE;
#pragma unroll
    for (int n = 0; n < DSTATE; ++n) {
        An[n] = -__expf(A_log[d * DSTATE + n]);
        h[n]  = Hin[base + n];
    }
    const float Dpd = Dp[d];

    for (int l = c * CHUNK; l < c * CHUNK + CHUNK; ++l) {
        float dlt = delta[(long)l * DINNER + d];
        float xv  = xconv[(long)l * DINNER + d];
        float dx  = dlt * xv;
        const f32x4* bc = (const f32x4*)(xdbl + (long)l * XPROJ + DTRANK);
        float Bv[DSTATE], Cv[DSTATE];
        ((f32x4*)Bv)[0] = bc[0]; ((f32x4*)Bv)[1] = bc[1];
        ((f32x4*)Bv)[2] = bc[2]; ((f32x4*)Bv)[3] = bc[3];
        ((f32x4*)Cv)[0] = bc[4]; ((f32x4*)Cv)[1] = bc[5];
        ((f32x4*)Cv)[2] = bc[6]; ((f32x4*)Cv)[3] = bc[7];

        float yv = 0.f;
#pragma unroll
        for (int n = 0; n < DSTATE; ++n) {
            float dA = __expf(dlt * An[n]);
            h[n] = dA * h[n] + dx * Bv[n];
            yv  += h[n] * Cv[n];
        }
        float zv  = xz[(long)l * (2 * DINNER) + DINNER + d];
        float sz  = zv / (1.f + __expf(-zv));          // SiLU(z)
        float out = (yv + Dpd * xv) * sz;
        y_bf[(long)l * DINNER + d] = f2bf(out);
    }
}

// ---------------------------------------------------------------- launcher
extern "C" void kernel_launch(void* const* d_in, const int* in_sizes, int n_in,
                              void* d_out, int out_size, void* d_ws, size_t ws_size,
                              hipStream_t stream) {
    (void)in_sizes; (void)n_in; (void)out_size; (void)ws_size;

    // inputs (all fp32), setup_inputs() order.  Batch 0 (hidden_states) is dead:
    // the scan is per-batch elementwise and only y[1] (from adjs) reaches the output.
    const float* adjs  = (const float*)d_in[1];   // [2048,1024]
    const float* ipw   = (const float*)d_in[2];   // [2048,1024]
    const float* cw    = (const float*)d_in[3];   // [1024,1,4]
    const float* cb    = (const float*)d_in[4];   // [1024]
    const float* xpw   = (const float*)d_in[5];   // [96,1024]
    const float* dpw   = (const float*)d_in[6];   // [1024,64]
    const float* dpb   = (const float*)d_in[7];   // [1024]
    const float* A_log = (const float*)d_in[8];   // [1024,16]
    const float* Dpar  = (const float*)d_in[9];   // [1024]
    const float* opw   = (const float*)d_in[10];  // [1024,1024]
    float* out = (float*)d_out;                   // [2048,1024]

    // workspace carve-out (256B aligned)
    char* w = (char*)d_ws;
    auto carve = [&](size_t bytes) -> char* {
        char* p = w;
        w += (bytes + 255) & ~(size_t)255;
        return p;
    };
    unsigned short* adjs_bf  = (unsigned short*)carve((size_t)LSEQ * DMODEL * 2);
    unsigned short* ipw_bf   = (unsigned short*)carve((size_t)2 * DINNER * DMODEL * 2);
    unsigned short* xpw_bf   = (unsigned short*)carve((size_t)XPROJ * DINNER * 2);
    unsigned short* dpw_bf   = (unsigned short*)carve((size_t)DINNER * DTRANK * 2);
    unsigned short* opw_bf   = (unsigned short*)carve((size_t)DMODEL * DINNER * 2);
    float*          xz       = (float*)carve((size_t)LSEQ * 2 * DINNER * 4);
    float*          xconv    = (float*)carve((size_t)LSEQ * DINNER * 4);
    unsigned short* xconv_bf = (unsigned short*)carve((size_t)LSEQ * DINNER * 2);
    float*          xdbl     = (float*)carve((size_t)LSEQ * XPROJ * 4);
    unsigned short* dt_bf    = (unsigned short*)carve((size_t)LSEQ * DTRANK * 2);
    float*          delta    = (float*)carve((size_t)LSEQ * DINNER * 4);
    unsigned short* y_bf     = (unsigned short*)carve((size_t)LSEQ * DINNER * 2);
    float*          Pbuf     = (float*)carve((size_t)NCHUNK * DINNER * DSTATE * 4);
    float*          Hend     = (float*)carve((size_t)NCHUNK * DINNER * DSTATE * 4);
    float*          Hin      = (float*)carve((size_t)NCHUNK * DINNER * DSTATE * 4);

    const int TB = 256;
    auto blocks = [](long n, int tb) { return (unsigned)((n + tb - 1) / tb); };

    // 1) fp32 -> bf16 conversions
    f32_to_bf16_kernel<<<blocks((long)LSEQ * DMODEL, TB), TB, 0, stream>>>(adjs, adjs_bf, (long)LSEQ * DMODEL);
    f32_to_bf16_kernel<<<blocks((long)2 * DINNER * DMODEL, TB), TB, 0, stream>>>(ipw, ipw_bf, (long)2 * DINNER * DMODEL);
    f32_to_bf16_kernel<<<blocks((long)XPROJ * DINNER, TB), TB, 0, stream>>>(xpw, xpw_bf, (long)XPROJ * DINNER);
    f32_to_bf16_kernel<<<blocks((long)DINNER * DTRANK, TB), TB, 0, stream>>>(dpw, dpw_bf, (long)DINNER * DTRANK);
    f32_to_bf16_kernel<<<blocks((long)DMODEL * DINNER, TB), TB, 0, stream>>>(opw, opw_bf, (long)DMODEL * DINNER);

    // 2) in_proj: xz[2048,2048] = adjs x ipw^T   (LDS/async WMMA GEMM)
    {
        dim3 grid((2 * DINNER) / BN, LSEQ / BM);   // 16 x 32
        gemm_lds_wmma<<<grid, 256, 0, stream>>>(adjs_bf, ipw_bf, xz, DMODEL, 2 * DINNER);
    }

    // 3) depthwise conv + SiLU
    conv_silu_kernel<<<blocks((long)LSEQ * DINNER, TB), TB, 0, stream>>>(xz, cw, cb, xconv, xconv_bf);

    // 4) x_proj: x_dbl[2048,96] = xconv x xpw^T  (direct WMMA GEMM, N=96)
    {
        int tiles = (LSEQ / 32) * (XPROJ / 32);
        gemm_wmma_bf16<<<blocks((long)tiles * 32, TB), TB, 0, stream>>>(
            xconv_bf, xpw_bf, xdbl, LSEQ, XPROJ, DINNER, XPROJ);
    }

    // 5) dt_low -> bf16; dt_proj GEMM (K=64); softplus(+bias)
    extract_dt_kernel<<<blocks((long)LSEQ * DTRANK, TB), TB, 0, stream>>>(xdbl, dt_bf);
    {
        int tiles = (LSEQ / 32) * (DINNER / 32);
        gemm_wmma_bf16<<<blocks((long)tiles * 32, TB), TB, 0, stream>>>(
            dt_bf, dpw_bf, delta, LSEQ, DINNER, DTRANK, DINNER);
    }
    softplus_bias_kernel<<<blocks((long)LSEQ * DINNER, TB), TB, 0, stream>>>(delta, dpb);

    // 6) chunked parallel scan (32 chunks x 64 steps) + fused gating
    scan_chunk_local<<<blocks((long)NCHUNK * DINNER, TB), TB, 0, stream>>>(
        delta, xconv, xdbl, A_log, Pbuf, Hend);
    scan_chunk_prefix<<<blocks((long)DINNER * DSTATE, TB), TB, 0, stream>>>(Pbuf, Hend, Hin);
    scan_chunk_apply<<<blocks((long)NCHUNK * DINNER, TB), TB, 0, stream>>>(
        delta, xconv, xdbl, xz, A_log, Dpar, Hin, y_bf);

    // 7) out_proj: out[2048,1024] = y x opw^T    (LDS/async WMMA GEMM)
    {
        dim3 grid(DMODEL / BN, LSEQ / BM);         // 8 x 32
        gemm_lds_wmma<<<grid, 256, 0, stream>>>(y_bf, opw_bf, out, DINNER, DMODEL);
    }
}
